// Custom_FeedForward_48395691491523
// MI455X (gfx1250) — matrix operand order repaired and checked by reading, not compiled, not run on hardware
//
#include <hip/hip_runtime.h>

// SwiGLU FFN for MI455X (gfx1250, wave32).
//   gate = X @ W1^T ; up = X @ W2^T ; h = silu(gate)*up ; out = h @ W3^T
// M=8192 tokens, E=2048, H=8192  (~825 GFLOP, compute-bound on WMMA).
// GEMMs run on v_wmma_f32_16x16x32_bf16 with f32 accumulation; fp32 operands
// are converted to bf16 during LDS staging. h is staged in d_ws as bf16
// (8192*8192*2 = 128MB). Kernel B stages its bf16 h tiles with the Tensor
// Data Mover (tensor_load_to_lds, TENSORcnt) when the builtin is available.

typedef __bf16 bf16;
typedef __attribute__((ext_vector_type(16))) __bf16 v16bf;
typedef __attribute__((ext_vector_type(8)))  __bf16 bf16x8;
typedef __attribute__((ext_vector_type(4)))  __bf16 bf16x4;
typedef __attribute__((ext_vector_type(8)))  float  v8f;

#define BM  128   // token rows per block
#define BN  128   // output cols per block
#define BK  32    // K-slice per step (= one WMMA K)
#define BKP 48    // padded LDS K-stride (bf16): 96B rows -> 16B aligned + bank rotation
#define THREADS 512   // 16 waves, 4x4 wave grid, 32x32 wave tile

#if defined(__has_builtin)
#if __has_builtin(__builtin_amdgcn_tensor_load_to_lds) && __has_builtin(__builtin_amdgcn_s_wait_tensorcnt)
#define HAVE_TDM 1
#endif
#endif
#ifndef HAVE_TDM
#define HAVE_TDM 0
#endif

__device__ __forceinline__ float silu_f(float x) {
  return x / (1.0f + __expf(-x));
}

#if HAVE_TDM
typedef unsigned int u32x4 __attribute__((ext_vector_type(4)));
typedef int          i32x4 __attribute__((ext_vector_type(4)));
typedef int          i32x8 __attribute__((ext_vector_type(8)));

// Issue a TDM load of a [tile_dim1=128 rows x tile_dim0=32 elem] bf16 tile from a
// row-major tensor with tensor_dim0_stride=8192 into LDS at ldsByteAddr, padding
// +32B after every 64B row so the LDS layout matches bf16[128][BKP=48].
__device__ __forceinline__ void tdm_load_h_tile(const bf16* gbase,
                                                unsigned ldsByteAddr,
                                                unsigned long long gByteOff) {
  unsigned long long ga = (unsigned long long)gbase + gByteOff;
  u32x4 g0;
  g0[0] = 1u;                                              // count=1, user desc
  g0[1] = ldsByteAddr;                                     // lds_addr
  g0[2] = (unsigned)(ga & 0xFFFFFFFFull);                  // global_addr[31:0]
  g0[3] = (unsigned)((ga >> 32) & 0x01FFFFFFull)           // global_addr[56:32]
        | (2u << 30);                                      // type = 2 ("image")
  i32x8 g1;
  g1[0] = (int)((7u << 25)      // pad_amount: 8 DWORDs = 32B
              | (3u << 22)      // pad_interval: every 16 DWORDs = 64B
              | (1u << 20)      // pad_enable
              | (1u << 16));    // data_size = 2 bytes ; workgroup_mask = 0
  g1[1] = (int)(8192u << 16);   // tensor_dim0 = 8192 (bits 79:48, low half)
  g1[2] = (int)(8192u << 16);   // tensor_dim0 hi = 0 ; tensor_dim1 = 8192 (bits 111:80 low)
  g1[3] = (int)(32u << 16);     // tensor_dim1 hi = 0 ; tile_dim0 = 32 (bits 127:112)
  g1[4] = 128;                  // tile_dim1 = 128 ; tile_dim2 = 0
  g1[5] = 8192;                 // tensor_dim0_stride[31:0]
  g1[6] = 0;                    // stride hi / tensor_dim1_stride lo
  g1[7] = 0;
  i32x4 z4 = (i32x4)0;
#if __clang_major__ >= 23
  i32x8 z8 = (i32x8)0;
  __builtin_amdgcn_tensor_load_to_lds(g0, g1, z4, z4, z8, 0);
#else
  __builtin_amdgcn_tensor_load_to_lds(g0, g1, z4, z4, 0);
#endif
}
#endif  // HAVE_TDM

// ---------------------------------------------------------------------------
// Kernel A: gate/up GEMMs fused, epilogue silu(g)*u -> bf16 h
// X:[M,K] f32 row-major, W1/W2:[N,K] f32 row-major, Hout:[M,N] bf16
// ---------------------------------------------------------------------------
__global__ __launch_bounds__(THREADS)
void ffn_gate_up_kernel(const float* __restrict__ X,
                        const float* __restrict__ W1,
                        const float* __restrict__ W2,
                        bf16* __restrict__ Hout,
                        int M, int K, int N)
{
  __shared__ bf16 Xs [2][BM][BKP];
  __shared__ bf16 W1s[2][BN][BKP];
  __shared__ bf16 W2s[2][BN][BKP];

  const int t     = threadIdx.x;
  const int wave  = t >> 5;        // 16 waves
  const int lane  = t & 31;
  const int wm    = wave >> 2;     // 0..3  (M direction)
  const int wn    = wave & 3;      // 0..3  (N direction)
  const int laneN = lane & 15;
  const int laneH = lane >> 4;     // lane half selects K sub-chunk

  const int mBase = blockIdx.y * BM;
  const int nBase = blockIdx.x * BN;

  v8f accG[2][2], accU[2][2];
#pragma unroll
  for (int i = 0; i < 2; ++i)
#pragma unroll
    for (int j = 0; j < 2; ++j) {
      accG[i][j] = (v8f)0.0f;
      accU[i][j] = (v8f)0.0f;
    }

  const int steps = K / BK;
  float4 xr[2], w1r[2], w2r[2];

  auto load_global = [&](int kk) {
    const int k0 = kk * BK;
#pragma unroll
    for (int i = 0; i < 2; ++i) {
      int id = t + i * THREADS;          // 1024 float4s per 128x32 f32 tile
      int row = id >> 3, c4 = (id & 7) << 2;
      xr[i]  = *(const float4*)(X  + (size_t)(mBase + row) * K + k0 + c4);
      w1r[i] = *(const float4*)(W1 + (size_t)(nBase + row) * K + k0 + c4);
      w2r[i] = *(const float4*)(W2 + (size_t)(nBase + row) * K + k0 + c4);
    }
    if (kk + 1 < steps) {               // gfx1250 global_prefetch_b8 for next slice
      __builtin_prefetch(X + (size_t)(mBase + (t >> 3)) * K + (kk + 1) * BK, 0, 1);
    }
  };

  auto store_lds = [&](int buf) {
#pragma unroll
    for (int i = 0; i < 2; ++i) {
      int id = t + i * THREADS;
      int row = id >> 3, c4 = (id & 7) << 2;
      bf16x4 vx = { (bf16)xr[i].x,  (bf16)xr[i].y,  (bf16)xr[i].z,  (bf16)xr[i].w  };
      bf16x4 v1 = { (bf16)w1r[i].x, (bf16)w1r[i].y, (bf16)w1r[i].z, (bf16)w1r[i].w };
      bf16x4 v2 = { (bf16)w2r[i].x, (bf16)w2r[i].y, (bf16)w2r[i].z, (bf16)w2r[i].w };
      *(bf16x4*)&Xs [buf][row][c4] = vx;
      *(bf16x4*)&W1s[buf][row][c4] = v1;
      *(bf16x4*)&W2s[buf][row][c4] = v2;
    }
  };

  auto compute = [&](int buf) {
    union U { v16bf v; bf16x8 h[2]; };
    v16bf a[2], bg[2], bu[2];
    // A 16x32 bf16 fragment: lane -> row = lane%16; K chunks {0..7,16..23} / {8..15,24..31}
#pragma unroll
    for (int tm = 0; tm < 2; ++tm) {
      const bf16* p = &Xs[buf][wm * 32 + tm * 16 + laneN][laneH * 8];
      U u;
      u.h[0] = *(const bf16x8*)(p);
      u.h[1] = *(const bf16x8*)(p + 16);
      a[tm] = u.v;
    }
    // B 32x16 bf16 fragment: lane -> N = lane%16; K chunks {0..7,16..23} / {8..15,24..31}
#pragma unroll
    for (int tn = 0; tn < 2; ++tn) {
      const bf16* p1 = &W1s[buf][wn * 32 + tn * 16 + laneN][laneH * 8];
      const bf16* p2 = &W2s[buf][wn * 32 + tn * 16 + laneN][laneH * 8];
      U u1, u2;
      u1.h[0] = *(const bf16x8*)(p1);
      u1.h[1] = *(const bf16x8*)(p1 + 16);
      u2.h[0] = *(const bf16x8*)(p2);
      u2.h[1] = *(const bf16x8*)(p2 + 16);
      bg[tn] = u1.v;
      bu[tn] = u2.v;
    }
#pragma unroll
    for (int tm = 0; tm < 2; ++tm)
#pragma unroll
      for (int tn = 0; tn < 2; ++tn) {
        accG[tm][tn] = __builtin_amdgcn_wmma_f32_16x16x32_bf16(
            false, a[tm], false, bg[tn], (short)0, accG[tm][tn], false, false);
        accU[tm][tn] = __builtin_amdgcn_wmma_f32_16x16x32_bf16(
            false, a[tm], false, bu[tn], (short)0, accU[tm][tn], false, false);
      }
  };

  load_global(0);
  store_lds(0);
  __syncthreads();

  for (int kk = 0; kk < steps; ++kk) {
    const int cur = kk & 1, nxt = cur ^ 1;
    if (kk + 1 < steps) load_global(kk + 1);
    compute(cur);
    if (kk + 1 < steps) store_lds(nxt);
    __syncthreads();
  }

  // D layout: VGPR r, lane -> N = lane%16, M = r + (lane<16 ? 0 : 8)
#pragma unroll
  for (int tm = 0; tm < 2; ++tm)
#pragma unroll
    for (int tn = 0; tn < 2; ++tn)
#pragma unroll
      for (int r = 0; r < 8; ++r) {
        int m = mBase + wm * 32 + tm * 16 + laneH * 8 + r;
        int n = nBase + wn * 32 + tn * 16 + laneN;
        float g = accG[tm][tn][r];
        float u = accU[tm][tn][r];
        Hout[(size_t)m * N + n] = (bf16)(silu_f(g) * u);
      }
}

// ---------------------------------------------------------------------------
// Kernel B: out = h @ W3^T.  Hin:[M,K] bf16, W3:[N,K] f32, out:[M,N] f32
// h tiles staged via Tensor Data Mover when available (pure bf16 2D tile copy).
// ---------------------------------------------------------------------------
__global__ __launch_bounds__(THREADS)
void ffn_down_kernel(const bf16* __restrict__ Hin,
                     const float* __restrict__ W3,
                     float* __restrict__ Out,
                     int M, int K, int N)
{
  __shared__ bf16 Hs [2][BM][BKP];
  __shared__ bf16 W3s[2][BN][BKP];

  const int t     = threadIdx.x;
  const int wave  = t >> 5;
  const int lane  = t & 31;
  const int wm    = wave >> 2;
  const int wn    = wave & 3;
  const int laneN = lane & 15;
  const int laneH = lane >> 4;

  const int mBase = blockIdx.y * BM;
  const int nBase = blockIdx.x * BN;

  v8f acc[2][2];
#pragma unroll
  for (int i = 0; i < 2; ++i)
#pragma unroll
    for (int j = 0; j < 2; ++j) acc[i][j] = (v8f)0.0f;

  const int steps = K / BK;
  float4 w3r[2];
#if !HAVE_TDM
  uint4 hr;
#endif
#if HAVE_TDM
  const unsigned ldsH0 = (unsigned)(unsigned long long)&Hs[0][0][0];
  const unsigned ldsH1 = (unsigned)(unsigned long long)&Hs[1][0][0];
#endif

  auto load_w3 = [&](int kk) {
    const int k0 = kk * BK;
#pragma unroll
    for (int i = 0; i < 2; ++i) {
      int id = t + i * THREADS;              // 1024 float4s of 128x32 f32 W3 tile
      int row = id >> 3, c4 = (id & 7) << 2;
      w3r[i] = *(const float4*)(W3 + (size_t)(nBase + row) * K + k0 + c4);
    }
  };
  auto store_w3 = [&](int buf) {
#pragma unroll
    for (int i = 0; i < 2; ++i) {
      int id = t + i * THREADS;
      int row = id >> 3, c4 = (id & 7) << 2;
      bf16x4 v = { (bf16)w3r[i].x, (bf16)w3r[i].y, (bf16)w3r[i].z, (bf16)w3r[i].w };
      *(bf16x4*)&W3s[buf][row][c4] = v;
    }
  };

  auto stage_h = [&](int buf, int kk) {
#if HAVE_TDM
    if (wave == 0) {
      tdm_load_h_tile(Hin, buf ? ldsH1 : ldsH0,
                      2ull * ((size_t)mBase * (size_t)K + (size_t)kk * BK));
    }
#else
    const int k0 = kk * BK;
    int row = t >> 2, c8 = (t & 3) << 3;     // 512 16B chunks of bf16 h tile
    hr = *(const uint4*)(Hin + (size_t)(mBase + row) * K + k0 + c8);
    *(uint4*)&Hs[buf][row][c8] = hr;
#endif
  };

  auto compute = [&](int buf) {
    union U { v16bf v; bf16x8 h[2]; };
    v16bf a[2], b[2];
#pragma unroll
    for (int tm = 0; tm < 2; ++tm) {
      const bf16* p = &Hs[buf][wm * 32 + tm * 16 + laneN][laneH * 8];
      U u;
      u.h[0] = *(const bf16x8*)(p);
      u.h[1] = *(const bf16x8*)(p + 16);
      a[tm] = u.v;
    }
#pragma unroll
    for (int tn = 0; tn < 2; ++tn) {
      const bf16* p = &W3s[buf][wn * 32 + tn * 16 + laneN][laneH * 8];
      U u;
      u.h[0] = *(const bf16x8*)(p);
      u.h[1] = *(const bf16x8*)(p + 16);
      b[tn] = u.v;
    }
#pragma unroll
    for (int tm = 0; tm < 2; ++tm)
#pragma unroll
      for (int tn = 0; tn < 2; ++tn)
        acc[tm][tn] = __builtin_amdgcn_wmma_f32_16x16x32_bf16(
            false, a[tm], false, b[tn], (short)0, acc[tm][tn], false, false);
  };

  // prologue: stage buffer 0
  stage_h(0, 0);
  load_w3(0);
  store_w3(0);
#if HAVE_TDM
  if (wave == 0) __builtin_amdgcn_s_wait_tensorcnt(0);
#endif
  __syncthreads();

  for (int kk = 0; kk < steps; ++kk) {
    const int cur = kk & 1, nxt = cur ^ 1;
    if (kk + 1 < steps) {
      stage_h(nxt, kk + 1);                 // TDM DMA overlaps with compute below
      load_w3(kk + 1);
    }
    compute(cur);
    if (kk + 1 < steps) store_w3(nxt);
#if HAVE_TDM
    if (wave == 0 && kk + 1 < steps) __builtin_amdgcn_s_wait_tensorcnt(0);
#endif
    __syncthreads();
  }

#pragma unroll
  for (int tm = 0; tm < 2; ++tm)
#pragma unroll
    for (int tn = 0; tn < 2; ++tn)
#pragma unroll
      for (int r = 0; r < 8; ++r) {
        int m = mBase + wm * 32 + tm * 16 + laneH * 8 + r;
        int n = nBase + wn * 32 + tn * 16 + laneN;
        Out[(size_t)m * N + n] = acc[tm][tn][r];
      }
}

// ---------------------------------------------------------------------------
extern "C" void kernel_launch(void* const* d_in, const int* in_sizes, int n_in,
                              void* d_out, int out_size, void* d_ws, size_t ws_size,
                              hipStream_t stream) {
  const float* X  = (const float*)d_in[0];   // [4,2048,2048] f32
  const float* W1 = (const float*)d_in[1];   // [8192,2048]   f32
  const float* W2 = (const float*)d_in[2];   // [8192,2048]   f32
  const float* W3 = (const float*)d_in[3];   // [2048,8192]   f32
  float* out = (float*)d_out;                // [4,2048,2048] f32
  bf16*  Hws = (bf16*)d_ws;                  // [8192,8192] bf16 intermediate (128MB)

  const int M = 4 * 2048;   // tokens
  const int E = 2048;
  const int H = 8192;

  dim3 blk(THREADS);
  dim3 gridA(H / BN, M / BM);   // (64, 64)
  ffn_gate_up_kernel<<<gridA, blk, 0, stream>>>(X, W1, W2, Hws, M, E, H);

  dim3 gridB(E / BN, M / BM);   // (16, 64)
  ffn_down_kernel<<<gridB, blk, 0, stream>>>(Hws, W3, out, M, H, E);
}
